// seq2seq_74534862455185
// MI455X (gfx1250) — compile-verified, matrix operand order checked by reading, not run on hardware
//
#include <hip/hip_runtime.h>

#define S_LEN  256   // encoder sequence length
#define BATCH  16
#define EMB    256
#define HID    256
#define GATES  1024  // 4*HID, PyTorch gate order i,f,g,o
#define TSTEPS 128
#define NLAYER 2

typedef __attribute__((ext_vector_type(16))) __bf16 bf16x16;
typedef __attribute__((ext_vector_type(8)))  float  f32x8;

union FragU { bf16x16 v; uint4 q[2]; };

__device__ __forceinline__ unsigned short f2bf(float f) {
  unsigned u = __float_as_uint(f);
  u += 0x7FFFu + ((u >> 16) & 1u);      // round-to-nearest-even
  return (unsigned short)(u >> 16);
}

__device__ __forceinline__ float sigm(float x) {
  return 1.0f / (1.0f + __expf(-x));
}

__device__ __forceinline__ f32x8 wmma_bf16(bf16x16 a, bf16x16 b, f32x8 c) {
  return __builtin_amdgcn_wmma_f32_16x16x32_bf16(
      /*neg_a=*/false, a, /*neg_b=*/false, b,
      /*c_mod=*/(short)0, c, /*reuse_a=*/false, /*reuse_b=*/false);
}

// Load a 16x32 bf16 A/B fragment from a row-major [*, ld] tile.
// Lane L holds row (L&15); halves at K offsets (L>=16)*8 and 16+(L>=16)*8.
__device__ __forceinline__ bf16x16 load_frag(const unsigned short* base, int ld,
                                             int k0, int lane) {
  FragU f;
  const unsigned short* p = base + (lane & 15) * ld + k0 + ((lane >> 4) << 3);
  f.q[0] = *(const uint4*)(p);
  f.q[1] = *(const uint4*)(p + 16);
  return f.v;
}

// ---------------------------------------------------------------- utilities
__global__ void cvt_bf16(const float* __restrict__ src,
                         unsigned short* __restrict__ dst, int n) {
  int i = blockIdx.x * blockDim.x + threadIdx.x;
  if (i < n) dst[i] = f2bf(src[i]);
}

__global__ void bias_comb(const float* __restrict__ a, const float* __restrict__ b,
                          float* __restrict__ o, int n) {
  int i = blockIdx.x * blockDim.x + threadIdx.x;
  if (i < n) o[i] = a[i] + b[i];
}

// X[row, e] = bf16(embed_table[source[row], e]); row = s*B + b
__global__ void embed_gather(const float* __restrict__ table,
                             const int* __restrict__ src,
                             unsigned short* __restrict__ X) {
  int row = blockIdx.x;
  int e = threadIdx.x;
  int idx = src[row];
  X[(size_t)row * EMB + e] = f2bf(table[(size_t)idx * EMB + e]);
}

// ------------------------------------------------- batched input projection
// Y[m, n] = sum_k Xbf[m, k] * W[n, k] + bias[n];  M=4096, N=1024, K=256.
// 8 waves / block, one 16x16 output tile per wave; grid = 2048 blocks.
__global__ void __launch_bounds__(256)
inproj_gemm(const unsigned short* __restrict__ Xbf,   // [4096, 256] bf16
            const unsigned short* __restrict__ W,     // [1024, 256] bf16
            const float* __restrict__ bias,           // [1024]
            float* __restrict__ Y) {                  // [4096, 1024] f32
  const int lane = threadIdx.x & 31;
  const int wave = threadIdx.x >> 5;
  const int tile = blockIdx.x * 8 + wave;
  const int tm = tile >> 6;    // 256 row tiles
  const int tn = tile & 63;    // 64 col tiles

  const unsigned short* abase = Xbf + (size_t)tm * 16 * EMB;
  const unsigned short* bbase =
      W + (size_t)(tn * 16 + (lane & 15)) * EMB + ((lane >> 4) << 3);

  f32x8 acc = {0.f, 0.f, 0.f, 0.f, 0.f, 0.f, 0.f, 0.f};
#pragma unroll
  for (int k = 0; k < 8; ++k) {
    bf16x16 fa = load_frag(abase, EMB, k * 32, lane);
    FragU fb;
    fb.q[0] = *(const uint4*)(bbase + k * 32);
    fb.q[1] = *(const uint4*)(bbase + k * 32 + 16);
    acc = wmma_bf16(fa, fb.v, acc);
  }
  float bv = bias[tn * 16 + (lane & 15)];
#pragma unroll
  for (int r = 0; r < 8; ++r)
    Y[((size_t)tm * 16 + r + ((lane >> 4) << 3)) * GATES + tn * 16 + (lane & 15)] =
        acc[r] + bv;
}

// ----------------------------------------------------- encoder layer scan
// One workgroup (16 wave32s) runs all 256 sequential steps.
// gates = XW[t] (precomputed x@Wih^T + biases) + h @ Whh^T (WMMA, bf16).
__global__ void __launch_bounds__(512)
lstm_scan(const float* __restrict__ XW,              // [S*B, G]
          const float* __restrict__ h0,              // [B, H]
          const float* __restrict__ c0,              // [B, H]
          const unsigned short* __restrict__ Whh,    // [G, H] bf16
          unsigned short* __restrict__ Hout,         // [S*B, H] bf16
          float* __restrict__ hfin) {                // [B, H] final h
  __shared__ __align__(16) unsigned short hbf[BATCH * HID];
  __shared__ float gates[BATCH * GATES];

  const int tid = threadIdx.x;
  const int lane = tid & 31;
  const int wave = tid >> 5;          // also the batch row this thread updates
  const int b = wave;
  const int hh0 = (tid & 31) * 8;

  float cv[8];                         // persistent cell state, 8 cells/thread
#pragma unroll
  for (int j = 0; j < 8; ++j) {
    cv[j] = c0[b * HID + hh0 + j];
    hbf[b * HID + hh0 + j] = f2bf(h0[b * HID + hh0 + j]);
  }
  __syncthreads();

  for (int t = 0; t < S_LEN; ++t) {
    // ---- recurrent matmul: each wave owns 4 of 64 gate column tiles ----
    bf16x16 fa[8];
#pragma unroll
    for (int k = 0; k < 8; ++k) fa[k] = load_frag(hbf, HID, k * 32, lane);

#pragma unroll
    for (int j = 0; j < 4; ++j) {
      const int g0 = (wave * 4 + j) * 16;
      f32x8 acc = {0.f, 0.f, 0.f, 0.f, 0.f, 0.f, 0.f, 0.f};
      const unsigned short* wb =
          Whh + (size_t)(g0 + (lane & 15)) * HID + ((lane >> 4) << 3);
#pragma unroll
      for (int k = 0; k < 8; ++k) {
        FragU fb;
        fb.q[0] = *(const uint4*)(wb + k * 32);
        fb.q[1] = *(const uint4*)(wb + k * 32 + 16);
        acc = wmma_bf16(fa[k], fb.v, acc);
      }
#pragma unroll
      for (int r = 0; r < 8; ++r)
        gates[(r + ((lane >> 4) << 3)) * GATES + g0 + (lane & 15)] = acc[r];
    }
    __syncthreads();

    // ---- elementwise cell update ----
    const float* xw = XW + ((size_t)t * BATCH + b) * GATES;
#pragma unroll
    for (int j = 0; j < 8; ++j) {
      const int hh = hh0 + j;
      float iv = sigm(gates[b * GATES + hh] + xw[hh]);
      float fv = sigm(gates[b * GATES + 256 + hh] + xw[256 + hh]);
      float gv = tanhf(gates[b * GATES + 512 + hh] + xw[512 + hh]);
      float ov = sigm(gates[b * GATES + 768 + hh] + xw[768 + hh]);
      float c = fv * cv[j] + iv * gv;
      cv[j] = c;
      float h = ov * tanhf(c);
      unsigned short hb = f2bf(h);
      hbf[b * HID + hh] = hb;
      Hout[((size_t)t * BATCH + b) * HID + hh] = hb;
      if (t == S_LEN - 1) hfin[b * HID + hh] = h;
    }
    __syncthreads();
  }
}

// --------------------------------------------------------- decoder scan
// 127 steps x 2 layers; input of step t is step t-1's top-layer h.
// Faithful bug: h AND c initialized from encoder final hidden states.
__global__ void __launch_bounds__(512)
lstm_decoder(const unsigned short* __restrict__ x0bf,  // [E] bf16
             const float* __restrict__ hidfin,         // [2, B, H]
             const unsigned short* __restrict__ Wih,   // [2, G, E] bf16
             const unsigned short* __restrict__ Whh,   // [2, G, H] bf16
             const float* __restrict__ bias,           // [2, G] bih+bhh
             float* __restrict__ out) {                // [T, 1, B, H]
  __shared__ __align__(16) unsigned short xbf[BATCH * EMB];
  __shared__ __align__(16) unsigned short hbf[NLAYER][BATCH * HID];
  __shared__ float gates[BATCH * GATES];

  const int tid = threadIdx.x;
  const int lane = tid & 31;
  const int wave = tid >> 5;
  const int b = wave;
  const int hh0 = (tid & 31) * 8;

  for (int e = tid; e < BATCH * HID; e += 512) out[e] = 0.f;  // preds[0] = 0

  float cv[NLAYER][8];
#pragma unroll
  for (int j = 0; j < 8; ++j) {
    float hl0 = hidfin[b * HID + hh0 + j];
    float hl1 = hidfin[BATCH * HID + b * HID + hh0 + j];
    cv[0][j] = hl0;
    cv[1][j] = hl1;
    hbf[0][b * HID + hh0 + j] = f2bf(hl0);
    hbf[1][b * HID + hh0 + j] = f2bf(hl1);
    xbf[b * EMB + hh0 + j] = x0bf[hh0 + j];   // tile x0 across batch
  }
  __syncthreads();

  for (int t = 0; t < TSTEPS - 1; ++t) {
#pragma unroll
    for (int l = 0; l < NLAYER; ++l) {
      const unsigned short* xs =
          (l == 0) ? ((t == 0) ? (const unsigned short*)xbf
                               : (const unsigned short*)hbf[1])
                   : (const unsigned short*)hbf[0];
      const unsigned short* hs = hbf[l];

      bf16x16 fa[8], fh[8];
#pragma unroll
      for (int k = 0; k < 8; ++k) {
        fa[k] = load_frag(xs, EMB, k * 32, lane);
        fh[k] = load_frag(hs, HID, k * 32, lane);
      }
      const unsigned short* wi = Wih + (size_t)l * GATES * EMB;
      const unsigned short* wh = Whh + (size_t)l * GATES * HID;

#pragma unroll
      for (int j = 0; j < 4; ++j) {
        const int g0 = (wave * 4 + j) * 16;
        f32x8 acc = {0.f, 0.f, 0.f, 0.f, 0.f, 0.f, 0.f, 0.f};
        const unsigned short* wbi =
            wi + (size_t)(g0 + (lane & 15)) * EMB + ((lane >> 4) << 3);
        const unsigned short* wbh =
            wh + (size_t)(g0 + (lane & 15)) * HID + ((lane >> 4) << 3);
#pragma unroll
        for (int k = 0; k < 8; ++k) {
          FragU fb;
          fb.q[0] = *(const uint4*)(wbi + k * 32);
          fb.q[1] = *(const uint4*)(wbi + k * 32 + 16);
          acc = wmma_bf16(fa[k], fb.v, acc);
        }
#pragma unroll
        for (int k = 0; k < 8; ++k) {
          FragU fb;
          fb.q[0] = *(const uint4*)(wbh + k * 32);
          fb.q[1] = *(const uint4*)(wbh + k * 32 + 16);
          acc = wmma_bf16(fh[k], fb.v, acc);
        }
#pragma unroll
        for (int r = 0; r < 8; ++r)
          gates[(r + ((lane >> 4) << 3)) * GATES + g0 + (lane & 15)] = acc[r];
      }
      __syncthreads();

      const float* bl = bias + l * GATES;
#pragma unroll
      for (int j = 0; j < 8; ++j) {
        const int hh = hh0 + j;
        float iv = sigm(gates[b * GATES + hh] + bl[hh]);
        float fv = sigm(gates[b * GATES + 256 + hh] + bl[256 + hh]);
        float gv = tanhf(gates[b * GATES + 512 + hh] + bl[512 + hh]);
        float ov = sigm(gates[b * GATES + 768 + hh] + bl[768 + hh]);
        float c = fv * cv[l][j] + iv * gv;
        cv[l][j] = c;
        float h = ov * tanhf(c);
        hbf[l][b * HID + hh] = f2bf(h);
        if (l == 1)
          out[((size_t)(t + 1)) * (BATCH * HID) + b * HID + hh] = h;
      }
      __syncthreads();
    }
  }
}

// -------------------------------------------------------------- launcher
extern "C" void kernel_launch(void* const* d_in, const int* in_sizes, int n_in,
                              void* d_out, int out_size, void* d_ws, size_t ws_size,
                              hipStream_t stream) {
  (void)in_sizes; (void)n_in; (void)out_size; (void)ws_size;
  const int*   source  = (const int*)  d_in[0];
  const float* tgt_emb = (const float*)d_in[2];   // [T,1,E]; only row 0 used
  const float* h0      = (const float*)d_in[3];   // [L,B,H]
  const float* c0      = (const float*)d_in[4];
  const float* table   = (const float*)d_in[5];
  const float* eWih    = (const float*)d_in[6];
  const float* eWhh    = (const float*)d_in[7];
  const float* eBih    = (const float*)d_in[8];
  const float* eBhh    = (const float*)d_in[9];
  const float* dWih    = (const float*)d_in[10];
  const float* dWhh    = (const float*)d_in[11];
  const float* dBih    = (const float*)d_in[12];
  const float* dBhh    = (const float*)d_in[13];
  float* out = (float*)d_out;

  // workspace bump allocator (~24.2 MB total)
  char* w = (char*)d_ws;
  auto alloc = [&](size_t bytes) {
    char* p = w; w += (bytes + 255) & ~(size_t)255; return p;
  };
  const size_t LW = (size_t)GATES * HID;          // 262144 elems per layer
  unsigned short* wihE = (unsigned short*)alloc(2 * LW * 2);
  unsigned short* whhE = (unsigned short*)alloc(2 * LW * 2);
  unsigned short* wihD = (unsigned short*)alloc(2 * LW * 2);
  unsigned short* whhD = (unsigned short*)alloc(2 * LW * 2);
  float* biasE = (float*)alloc(2 * GATES * 4);
  float* biasD = (float*)alloc(2 * GATES * 4);
  unsigned short* x0bf = (unsigned short*)alloc(EMB * 2);
  float* hidfin = (float*)alloc(2 * BATCH * HID * 4);
  unsigned short* Xbf  = (unsigned short*)alloc((size_t)S_LEN * BATCH * HID * 2);
  unsigned short* H0bf = (unsigned short*)alloc((size_t)S_LEN * BATCH * HID * 2);
  float* XW = (float*)alloc((size_t)S_LEN * BATCH * GATES * 4);

  const int nW = (int)(2 * LW);                   // 524288
  cvt_bf16<<<(nW + 255) / 256, 256, 0, stream>>>(eWih, wihE, nW);
  cvt_bf16<<<(nW + 255) / 256, 256, 0, stream>>>(eWhh, whhE, nW);
  cvt_bf16<<<(nW + 255) / 256, 256, 0, stream>>>(dWih, wihD, nW);
  cvt_bf16<<<(nW + 255) / 256, 256, 0, stream>>>(dWhh, whhD, nW);
  bias_comb<<<8, 256, 0, stream>>>(eBih, eBhh, biasE, 2 * GATES);
  bias_comb<<<8, 256, 0, stream>>>(dBih, dBhh, biasD, 2 * GATES);
  cvt_bf16<<<1, 256, 0, stream>>>(tgt_emb, x0bf, EMB);
  embed_gather<<<S_LEN * BATCH, EMB, 0, stream>>>(table, source, Xbf);

  // encoder layer 0: hoisted input projection, then sequential scan
  inproj_gemm<<<2048, 256, 0, stream>>>(Xbf, wihE, biasE, XW);
  lstm_scan<<<1, 512, 0, stream>>>(XW, h0, c0, whhE, H0bf, hidfin);
  // encoder layer 1 (XW buffer reused; per-step output only feeds dead attention,
  // dumped into the now-dead Xbf buffer — only hidfin[1] is live)
  inproj_gemm<<<2048, 256, 0, stream>>>(H0bf, wihE + LW, biasE + GATES, XW);
  lstm_scan<<<1, 512, 0, stream>>>(XW, h0 + BATCH * HID, c0 + BATCH * HID,
                                   whhE + LW, Xbf, hidfin + BATCH * HID);
  // decoder: 127 self-feeding steps, writes preds[1:] and zeroes preds[0]
  lstm_decoder<<<1, 512, 0, stream>>>(x0bf, hidfin, wihD, whhD, biasD, out);
}
